// SovereignLeviathan_62783831933696
// MI455X (gfx1250) — matrix operand order repaired and checked by reading, not compiled, not run on hardware
//
#include <hip/hip_runtime.h>

// ---------------------------------------------------------------------------
// MoE block forward for MI455X (gfx1250, wave32, bf16 WMMA 16x16x32 + TDM)
//   x[B,S] -> emb gather -> router top2 softmax -> dense-expert FFN(ReLU)
//   -> gate-combine -> head GEMM [4096,3072]x[3072,32000] + bias
// ---------------------------------------------------------------------------

typedef __bf16 bf16;
typedef __bf16 bf16x8  __attribute__((ext_vector_type(8)));
typedef __bf16 bf16x16 __attribute__((ext_vector_type(16)));
typedef float  f32x8   __attribute__((ext_vector_type(8)));
typedef unsigned int u32x4 __attribute__((ext_vector_type(4)));
typedef int          i32x4 __attribute__((ext_vector_type(4)));
typedef int          i32x8 __attribute__((ext_vector_type(8)));

#define NTOK   4096      // B*S tokens
#define DMODEL 768
#define NEXP   8
#define DFF    3072
#define NVOCAB 32000

#define KT_D   24        // 768/32  K-tiles for FFN GEMM
#define NT_FF  192       // 3072/16 N-tiles for FFN GEMM
#define KT_FF  96        // 3072/32 K-tiles for head GEMM
#define NT_V   2000      // 32000/16 N-tiles for head GEMM

// ---- workspace layout (bytes) ---------------------------------------------
static constexpr size_t OFF_TBF   = 0;                                   // [4096][768] bf16
static constexpr size_t SZ_TBF    = (size_t)NTOK * DMODEL * 2;
static constexpr size_t OFF_GATES = OFF_TBF + SZ_TBF;                    // [4096][8] f32
static constexpr size_t SZ_GATES  = (size_t)NTOK * NEXP * 4;
static constexpr size_t OFF_WPK   = OFF_GATES + SZ_GATES;                // packed exp_w bf16
static constexpr size_t SZ_WPK    = (size_t)NEXP * NT_FF * KT_D * 512 * 2;
static constexpr size_t OFF_MOE   = OFF_WPK + SZ_WPK;                    // [4096][3072] bf16
static constexpr size_t SZ_MOE    = (size_t)NTOK * DFF * 2;
static constexpr size_t OFF_HPK   = OFF_MOE + SZ_MOE;                    // packed head_w bf16
static constexpr size_t SZ_HPK    = (size_t)NT_V * KT_FF * 512 * 2;

// ---------------------------------------------------------------------------
// TDM: issue a 2D tensor_load_to_lds (this toolchain's 6-arg builtin:
// (u32x4 g0, i32x8 g1, i32x4 g2, i32x4 g3, i32x8 extra, i32 cpol)).
// D# bitfields per CDNA5 ISA ch.8: group0 = {count, lds_addr, global_addr,
// type=2}; group1 = {data_size=2B + LDS pad (every 16 dwords insert 4 dwords
// -> 80B row pitch for a 64B row), tensor dims, tile dims, dim0 stride}.
// Groups 2/3 (+extra) are unused for a 2D tile -> zeros.
// ---------------------------------------------------------------------------
__device__ __forceinline__ void tdm_load_2d(unsigned long long gaddr,
                                            unsigned lds_off,
                                            unsigned dim0, unsigned dim1,
                                            unsigned tile0, unsigned tile1,
                                            unsigned stride0) {
  u32x4 g0;
  g0[0] = 1u;                                            // count=1 (valid)
  g0[1] = lds_off;                                       // lds_addr
  g0[2] = (unsigned)(gaddr & 0xffffffffu);               // global_addr[31:0]
  g0[3] = (unsigned)((gaddr >> 32) & 0x01ffffffu)        // global_addr[56:32]
        | (2u << 30);                                    // type = 2 ("image")
  i32x8 g1;
  g1[0] = (int)((1u << 16)      // data_size = 1 -> 2 bytes
              | (1u << 20)      // pad_enable
              | (3u << 22)      // pad_interval = 16 dwords
              | (3u << 25));    // pad_amount   = 4 dwords
  g1[1] = (int)((dim0 & 0xffffu) << 16);                           // dim0 lo
  g1[2] = (int)(((dim0 >> 16) & 0xffffu) | ((dim1 & 0xffffu) << 16));
  g1[3] = (int)(((dim1 >> 16) & 0xffffu) | ((tile0 & 0xffffu) << 16));
  g1[4] = (int)(tile1 & 0xffffu);                        // tile_dim1; dim2=0
  g1[5] = (int)stride0;                                  // dim0_stride lo
  g1[6] = 0;
  g1[7] = 0;
  i32x4 z4 = {0, 0, 0, 0};                               // groups 2/3 unused
  i32x8 z8 = {0, 0, 0, 0, 0, 0, 0, 0};
  __builtin_amdgcn_tensor_load_to_lds(g0, g1, z4, z4, z8, 0);
}

// ---------------------------------------------------------------------------
// A-fragment loader from row-major global bf16 (16x32 bf16 A layout:
// lane L: row L&15; elems 0..7 = K (L>>4)*8.. ; elems 8..15 = K 16+(L>>4)*8..)
// ---------------------------------------------------------------------------
__device__ __forceinline__ bf16x16 load_a_frag(const bf16* __restrict__ A,
                                               int stride, int row0, int k0,
                                               int lane) {
  int r  = row0 + (lane & 15);
  int kh = (lane >> 4) * 8;
  const bf16* p = A + (size_t)r * stride + k0 + kh;
  bf16x8 lo = *(const bf16x8*)(p);
  bf16x8 hi = *(const bf16x8*)(p + 16);
  return __builtin_shufflevector(lo, hi, 0, 1, 2, 3, 4, 5, 6, 7,
                                         8, 9, 10, 11, 12, 13, 14, 15);
}

// Same fragment, but from an 80-byte-pitch LDS tile written by the TDM.
__device__ __forceinline__ bf16x16 load_a_frag_lds(const bf16* sA, int row,
                                                   int lane) {
  int r  = row + (lane & 15);
  int kh = (lane >> 4) * 8;
  const char* p = (const char*)sA + r * 80 + kh * 2;
  bf16x8 lo = *(const bf16x8*)(p);
  bf16x8 hi = *(const bf16x8*)(p + 32);                  // +16 elements
  return __builtin_shufflevector(lo, hi, 0, 1, 2, 3, 4, 5, 6, 7,
                                         8, 9, 10, 11, 12, 13, 14, 15);
}

// ---------------------------------------------------------------------------
// Pack f32 row-major [nmat][nkt*32][ncols] into WMMA B-fragment-native bf16
// tiles: tile (m,nt,kt) -> 512 bf16; lane L's 16 values are
// W[kt*32 + (L>>4)*8 + i + (i>=8 ? 8 : 0)][nt*16 + (L&15)]  -> one 32B store per lane.
// ---------------------------------------------------------------------------
__global__ void pack_b_kernel(const float* __restrict__ W, bf16* __restrict__ O,
                              int ncols, int nkt, int ntn, int nmat) {
  long g    = (long)blockIdx.x * blockDim.x + threadIdx.x;
  int  lane = (int)(g & 31);
  long tile = g >> 5;
  long tiles_per_mat = (long)ntn * nkt;
  if (tile >= (long)nmat * tiles_per_mat) return;
  int  m  = (int)(tile / tiles_per_mat);
  long rr = tile - (long)m * tiles_per_mat;
  int  nt = (int)(rr / nkt);
  int  kt = (int)(rr - (long)nt * nkt);
  const float* Wm = W + (size_t)m * (size_t)nkt * 32 * ncols;
  int kbase = kt * 32 + (lane >> 4) * 8;
  int n     = nt * 16 + (lane & 15);
  bf16x16 v;
#pragma unroll
  for (int i = 0; i < 16; ++i)
    v[i] = (bf16)Wm[(size_t)(kbase + i + (i >= 8 ? 8 : 0)) * ncols + n];
  *(bf16x16*)(O + tile * 512 + (size_t)lane * 16) = v;
}

// ---------------------------------------------------------------------------
// Embedding gather + router (logits -> top-2 -> softmax -> dense gate row).
// ---------------------------------------------------------------------------
__global__ void embed_router_kernel(const int* __restrict__ x,
                                    const float* __restrict__ emb,
                                    const float* __restrict__ gw,   // [768][8]
                                    const float* __restrict__ gb,   // [8]
                                    bf16* __restrict__ tbf,         // [4096][768]
                                    float* __restrict__ gates) {    // [4096][8]
  int t   = blockIdx.x;
  int tid = threadIdx.x;
  int idx = x[t];
  const float* row = emb + (size_t)idx * DMODEL;

  float p[NEXP];
#pragma unroll
  for (int e = 0; e < NEXP; ++e) p[e] = 0.f;

  for (int d = tid; d < DMODEL; d += 256) {
    float v = row[d];
    tbf[(size_t)t * DMODEL + d] = (bf16)v;
#pragma unroll
    for (int e = 0; e < NEXP; ++e) p[e] += v * gw[d * NEXP + e];
  }

  __shared__ float sm[256 * NEXP];
#pragma unroll
  for (int e = 0; e < NEXP; ++e) sm[tid * NEXP + e] = p[e];
  __syncthreads();

  __shared__ float lg[NEXP];
  if (tid < NEXP) {
    float s = 0.f;
    for (int j = 0; j < 256; ++j) s += sm[j * NEXP + tid];
    lg[tid] = s + gb[tid];
  }
  __syncthreads();

  if (tid == 0) {
    float m1 = -3.4e38f, m2 = -3.4e38f;
    int   i1 = 0,        i2 = 0;
#pragma unroll
    for (int e = 0; e < NEXP; ++e) {
      float v = lg[e];
      if (v > m1)      { m2 = m1; i2 = i1; m1 = v; i1 = e; }
      else if (v > m2) { m2 = v;  i2 = e; }
    }
    float w1 = 1.f / (1.f + __expf(m2 - m1));
    float w2 = 1.f - w1;
#pragma unroll
    for (int e = 0; e < NEXP; ++e)
      gates[(size_t)t * NEXP + e] = (e == i1) ? w1 : ((e == i2) ? w2 : 0.f);
  }
}

// ---------------------------------------------------------------------------
// Dense-over-experts FFN + gated combine -> moe (bf16).
// Block = 16 tokens x 128 F-cols; 8 waves, one 16-wide N-tile each.
// ---------------------------------------------------------------------------
__global__ void moe_ffn_kernel(const bf16* __restrict__ tbf,
                               const float* __restrict__ gates,
                               const bf16* __restrict__ wpk,
                               const float* __restrict__ expb,
                               bf16* __restrict__ moe) {
  int m    = blockIdx.x;
  int fc   = blockIdx.y;
  int tid  = threadIdx.x;
  int lane = tid & 31;
  int wv   = tid >> 5;
  int nt   = fc * 8 + wv;

  __shared__ float g_s[16 * NEXP];
  if (tid < 16 * NEXP) g_s[tid] = gates[(size_t)m * 16 * NEXP + tid];
  __syncthreads();

  int col   = lane & 15;
  int f     = nt * 16 + col;
  int rbase = (lane >> 4) * 8;

  float mo[8];
#pragma unroll
  for (int r = 0; r < 8; ++r) mo[r] = 0.f;

  for (int e = 0; e < NEXP; ++e) {
    f32x8 acc = {};
    const bf16* bbase =
        wpk + (((size_t)e * NT_FF + nt) * KT_D) * 512 + (size_t)lane * 16;
#pragma unroll 4
    for (int kt = 0; kt < KT_D; ++kt) {
      bf16x16 a = load_a_frag(tbf, DMODEL, m * 16, kt * 32, lane);
      bf16x16 b = *(const bf16x16*)(bbase + (size_t)kt * 512);
      acc = __builtin_amdgcn_wmma_f32_16x16x32_bf16(
          false, a, false, b, (short)0, acc, false, false);
    }
    float bias = expb[e * DFF + f];
#pragma unroll
    for (int r = 0; r < 8; ++r) {
      float v = acc[r] + bias;
      v = v > 0.f ? v : 0.f;
      mo[r] += g_s[(rbase + r) * NEXP + e] * v;
    }
  }
#pragma unroll
  for (int r = 0; r < 8; ++r) {
    int tok = m * 16 + rbase + r;
    moe[(size_t)tok * DFF + f] = (bf16)mo[r];
  }
}

// ---------------------------------------------------------------------------
// Head GEMM: out[4096,32000] = moe_bf @ head_w_bf + head_b  (f32 out).
// Block = 64 rows x 256 cols; 8 waves (2x4), each wave owns 2x4 tiles.
// A tiles (64x32 bf16) are TDM-DMA'd into double-buffered LDS with 80B row
// pitch (bank-conflict-free fragment reads); B streams from packed global.
// ---------------------------------------------------------------------------
__global__ void head_kernel(const bf16* __restrict__ moe,
                            const bf16* __restrict__ hpk,
                            const float* __restrict__ hb,
                            float* __restrict__ out) {
  __shared__ __align__(16) bf16 sA[2][64 * 40];   // 2 x 64 rows x 80B pitch

  int mb   = blockIdx.x;            // 0..63   (64-row block)
  int nb   = blockIdx.y;            // 0..124  (256-col block)
  int tid  = threadIdx.x;
  int lane = tid & 31;
  int wv   = tid >> 5;
  int wm   = wv >> 2;               // 0..1 : 32-row half
  int wn   = wv & 3;                // 0..3 : 64-col quarter
  int nt0  = nb * 16 + wn * 4;      // first of 4 N-tiles for this wave

  unsigned long long abase = (unsigned long long)(uintptr_t)moe
                           + (unsigned long long)mb * 64 * DFF * 2;
  unsigned lds0 = (unsigned)(uintptr_t)(&sA[0][0]);
  unsigned lds1 = (unsigned)(uintptr_t)(&sA[1][0]);

  if (wv == 0)                      // prologue: DMA tile kt=0 into buf0
    tdm_load_2d(abase, lds0, DFF, 64, 32, 64, DFF);

  const bf16* bbase[4];
#pragma unroll
  for (int ni = 0; ni < 4; ++ni)
    bbase[ni] = hpk + ((size_t)(nt0 + ni) * KT_FF) * 512 + (size_t)lane * 16;

  f32x8 acc[2][4] = {};

  for (int kt = 0; kt < KT_FF; ++kt) {
    __syncthreads();                // consumers done with buf[(kt+1)&1]
    if (wv == 0) {
      if (kt + 1 < KT_FF) {
        unsigned ldsn = ((kt + 1) & 1) ? lds1 : lds0;
        tdm_load_2d(abase + (unsigned long long)(kt + 1) * 64,  // +32 elems
                    ldsn, DFF, 64, 32, 64, DFF);
        __builtin_amdgcn_s_wait_tensorcnt(1);   // tile kt finished (in-order)
      } else {
        __builtin_amdgcn_s_wait_tensorcnt(0);
      }
    }
    __syncthreads();                // publish buf[kt&1]

    const bf16* sa = &sA[kt & 1][0];
    bf16x16 a0 = load_a_frag_lds(sa, wm * 32 + 0,  lane);
    bf16x16 a1 = load_a_frag_lds(sa, wm * 32 + 16, lane);

    bf16x16 b[4];
#pragma unroll
    for (int ni = 0; ni < 4; ++ni) {
      if (kt + 8 < KT_FF)
        __builtin_prefetch(bbase[ni] + (size_t)(kt + 8) * 512);
      b[ni] = *(const bf16x16*)(bbase[ni] + (size_t)kt * 512);
    }
#pragma unroll
    for (int ni = 0; ni < 4; ++ni) {
      acc[0][ni] = __builtin_amdgcn_wmma_f32_16x16x32_bf16(
          false, a0, false, b[ni], (short)0, acc[0][ni], false, false);
      acc[1][ni] = __builtin_amdgcn_wmma_f32_16x16x32_bf16(
          false, a1, false, b[ni], (short)0, acc[1][ni], false, false);
    }
  }

  int col   = lane & 15;
  int rbase = (lane >> 4) * 8;
#pragma unroll
  for (int ni = 0; ni < 4; ++ni) {
    int n = (nt0 + ni) * 16 + col;
    float bias = hb[n];
#pragma unroll
    for (int mi = 0; mi < 2; ++mi) {
#pragma unroll
      for (int r = 0; r < 8; ++r) {
        int tok = mb * 64 + wm * 32 + mi * 16 + rbase + r;
        out[(size_t)tok * NVOCAB + n] = acc[mi][ni][r] + bias;
      }
    }
  }
}

// ---------------------------------------------------------------------------
extern "C" void kernel_launch(void* const* d_in, const int* in_sizes, int n_in,
                              void* d_out, int out_size, void* d_ws,
                              size_t ws_size, hipStream_t stream) {
  (void)in_sizes; (void)n_in; (void)out_size; (void)ws_size;

  const int*   x   = (const int*)d_in[0];
  const float* emb = (const float*)d_in[1];
  const float* gw  = (const float*)d_in[2];
  const float* gb  = (const float*)d_in[3];
  const float* ew  = (const float*)d_in[4];
  const float* eb  = (const float*)d_in[5];
  const float* hw  = (const float*)d_in[6];
  const float* hb  = (const float*)d_in[7];
  float* out = (float*)d_out;

  char* ws     = (char*)d_ws;
  bf16*  tbf   = (bf16*)(ws + OFF_TBF);
  float* gates = (float*)(ws + OFF_GATES);
  bf16*  wpk   = (bf16*)(ws + OFF_WPK);
  bf16*  moe   = (bf16*)(ws + OFF_MOE);
  bf16*  hpk   = (bf16*)(ws + OFF_HPK);

  {
    long thr = (long)NEXP * NT_FF * KT_D * 32;
    pack_b_kernel<<<(int)((thr + 255) / 256), 256, 0, stream>>>(
        ew, wpk, DFF, KT_D, NT_FF, NEXP);
  }
  {
    long thr = (long)NT_V * KT_FF * 32;
    pack_b_kernel<<<(int)((thr + 255) / 256), 256, 0, stream>>>(
        hw, hpk, NVOCAB, KT_FF, NT_V, 1);
  }

  embed_router_kernel<<<NTOK, 256, 0, stream>>>(x, emb, gw, gb, tbf, gates);

  moe_ffn_kernel<<<dim3(NTOK / 16, DFF / 128), 256, 0, stream>>>(
      tbf, gates, wpk, eb, moe);

  head_kernel<<<dim3(NTOK / 64, NVOCAB / 256), 256, 0, stream>>>(
      moe, hpk, hb, out);
}
